// SenseContrastLoss_67276367725127
// MI455X (gfx1250) — matrix operand order repaired and checked by reading, not compiled, not run on hardware
//
#include <hip/hip_runtime.h>
#include <hip/hip_bf16.h>

// Problem constants (match reference)
#define NROWS 8192
#define DIMS  1024
#define INV_T 14.285714285714286f   // 1/0.07

// Tile geometry
#define BM      256                 // block rows
#define BN      128                 // block cols
#define KSTEP   64                  // K halves staged per stage (2 WMMA k-steps)
#define NSTAGE  (DIMS / KSTEP)      // 16
#define ROWH    (KSTEP + 8)         // 72 halves per LDS row (36*lo mod 64 injective)
#define ROWB    (ROWH * 2)          // 144 bytes per LDS row

typedef __attribute__((ext_vector_type(16))) _Float16 v16h;
typedef __attribute__((ext_vector_type(8)))  _Float16 v8h;
typedef __attribute__((ext_vector_type(8)))  float    v8f;

// ---------------------------------------------------------------------------
// CDNA5 async global->LDS copy (ASYNCcnt-tracked), CDNA5 ISA
// ---------------------------------------------------------------------------
__device__ __forceinline__ void async_copy_b128(unsigned lds_off, const void* gsrc) {
  asm volatile("global_load_async_to_lds_b128 %0, %1, off"
               :: "v"(lds_off), "v"(gsrc)
               : "memory");
}

template <int N>
__device__ __forceinline__ void wait_async() {
#if __has_builtin(__builtin_amdgcn_s_wait_asynccnt)
  __builtin_amdgcn_s_wait_asynccnt(N);
#else
  asm volatile("s_wait_asynccnt %0" :: "i"(N) : "memory");
#endif
}

// ---------------------------------------------------------------------------
// Kernel 1: row-normalize fp32 vectors -> f16 matrix in workspace
// ---------------------------------------------------------------------------
__global__ __launch_bounds__(256)
void scl_normalize(const float* __restrict__ vecs, _Float16* __restrict__ anorm) {
  __shared__ float red[256];
  const int row = blockIdx.x;
  const int tid = threadIdx.x;
  const float* src = vecs + (size_t)row * DIMS;
  float4 x = *(const float4*)(src + tid * 4);
  red[tid] = x.x * x.x + x.y * x.y + x.z * x.z + x.w * x.w;
  __syncthreads();
  for (int s = 128; s > 0; s >>= 1) {
    if (tid < s) red[tid] += red[tid + s];
    __syncthreads();
  }
  const float rn = rsqrtf(red[0]);
  _Float16* dst = anorm + (size_t)row * DIMS + tid * 4;
  dst[0] = (_Float16)(x.x * rn);
  dst[1] = (_Float16)(x.y * rn);
  dst[2] = (_Float16)(x.z * rn);
  dst[3] = (_Float16)(x.w * rn);
}

// ---------------------------------------------------------------------------
// Kernel 2: zero global row accumulators (workspace is poisoned 0xAA)
// ---------------------------------------------------------------------------
__global__ void scl_zero(float* gSum, float* gPos, int* gCnt) {
  const int i = blockIdx.x * blockDim.x + threadIdx.x;
  if (i < NROWS) { gSum[i] = 0.f; gPos[i] = 0.f; gCnt[i] = 0; }
}

// ---------------------------------------------------------------------------
// Kernel 3: fused tile GEMM (WMMA f16 -> f32) + online reductions.
// 256x128 tiles; per-element symmetry: only j > i is consumed, contributing
// both row-i and mirrored col-j partials. Blocks entirely below the diagonal
// exit early. A/B tiles double-buffered in LDS via async global->LDS copies.
// ---------------------------------------------------------------------------
__global__ __launch_bounds__(256)
void scl_tiles(const _Float16* __restrict__ A, const int* __restrict__ labels,
               float* __restrict__ gSum, float* __restrict__ gPos,
               int* __restrict__ gCnt) {
  const int br = blockIdx.y;
  const int bc = blockIdx.x;
  if ((bc + 1) * BN <= br * BM) return;   // tile entirely below diagonal
  const int rowBase = br * BM;
  const int colBase = bc * BN;

  __shared__ __align__(16) _Float16 tileA[2][BM * ROWH];   // 2 x 36 KB
  __shared__ __align__(16) _Float16 tileB[2][BN * ROWH];   // 2 x 18 KB
  __shared__ float sSum[BM], sPos[BM];
  __shared__ int   sCnt[BM];
  __shared__ float cSum[BN], cPos[BN];
  __shared__ int   cCnt[BN];
  __shared__ int   lRow[BM], lCol[BN];

  const int tid = threadIdx.x;
  sSum[tid] = 0.f; sPos[tid] = 0.f; sCnt[tid] = 0;
  lRow[tid] = labels[rowBase + tid];
  if (tid < BN) {
    cSum[tid] = 0.f; cPos[tid] = 0.f; cCnt[tid] = 0;
    lCol[tid] = labels[colBase + tid];
  }

  const int wid   = tid >> 5;   // wave id 0..7 (wave32)
  const int lane  = tid & 31;
  const int waveM = wid >> 1;   // 0..3 -> 64-row strip
  const int waveN = wid & 1;    // 0..1 -> 64-col strip
  const int lo    = lane & 15;
  const int hi    = lane >> 4;

  const unsigned ldsA[2] = {(unsigned)(unsigned long long)(void*)&tileA[0][0],
                            (unsigned)(unsigned long long)(void*)&tileA[1][0]};
  const unsigned ldsB[2] = {(unsigned)(unsigned long long)(void*)&tileB[0][0],
                            (unsigned)(unsigned long long)(void*)&tileB[1][0]};

  // Stage issuance: A = BM x KSTEP halves (8 chunks/thread), B = BN x KSTEP
  // (4 chunks/thread); 16 B chunks, coalesced along K within each row.
  auto issue_stage = [&](int stage, int buf) {
    const int kb = stage * KSTEP;
#pragma unroll
    for (int q = 0; q < 8; ++q) {                   // A: 2048 chunks
      const int chunk = tid + q * 256;
      const int r = chunk >> 3, c = chunk & 7;
      async_copy_b128(ldsA[buf] + r * ROWB + c * 16,
                      A + (size_t)(rowBase + r) * DIMS + kb + c * 8);
    }
#pragma unroll
    for (int q = 0; q < 4; ++q) {                   // B: 1024 chunks
      const int chunk = tid + q * 256;
      const int r = chunk >> 3, c = chunk & 7;
      async_copy_b128(ldsB[buf] + r * ROWB + c * 16,
                      A + (size_t)(colBase + r) * DIMS + kb + c * 8);
    }
  };

  // f32 accumulators: 4 (M) x 4 (N) tiles of 16x16 per wave (64x64 wave tile)
  v8f acc[4][4];
#pragma unroll
  for (int i = 0; i < 4; ++i)
#pragma unroll
    for (int j = 0; j < 4; ++j) {
      v8f z = {0.f, 0.f, 0.f, 0.f, 0.f, 0.f, 0.f, 0.f};
      acc[i][j] = z;
    }

  issue_stage(0, 0);
  issue_stage(1, 1);

  for (int s = 0; s < NSTAGE; ++s) {
    // Async copies retire in order; <=12 outstanding leaves only next stage.
    if (s + 1 < NSTAGE) wait_async<12>(); else wait_async<0>();
    __syncthreads();

    const char* bufA = (const char*)&tileA[s & 1][0];
    const char* bufB = (const char*)&tileB[s & 1][0];

#pragma unroll
    for (int kk = 0; kk < KSTEP / 32; ++kk) {
      const int kob = kk * 64;  // byte offset of this 32-half k-step in a row
      // A fragment (16-bit 16x32): lane holds K = 8*hi..+7 and 16+8*hi..+7
      v16h af[4];
#pragma unroll
      for (int tm = 0; tm < 4; ++tm) {
        const char* pr = bufA + (waveM * 64 + tm * 16 + lo) * ROWB + kob + 16 * hi;
        v8h a0 = *(const v8h*)(pr);
        v8h a1 = *(const v8h*)(pr + 32);
        af[tm] = __builtin_shufflevector(a0, a1, 0, 1, 2, 3, 4, 5, 6, 7,
                                         8, 9, 10, 11, 12, 13, 14, 15);
      }
      // B fragment (16-bit 32x16): B = A^T -> K = 8*hi..+7 and 16+8*hi..+7
      v16h bf[4];
#pragma unroll
      for (int tn = 0; tn < 4; ++tn) {
        const char* pc = bufB + (waveN * 64 + tn * 16 + lo) * ROWB + kob + 16 * hi;
        v8h b0 = *(const v8h*)(pc);
        v8h b1 = *(const v8h*)(pc + 32);
        bf[tn] = __builtin_shufflevector(b0, b1, 0, 1, 2, 3, 4, 5, 6, 7,
                                         8, 9, 10, 11, 12, 13, 14, 15);
      }
#pragma unroll
      for (int tm = 0; tm < 4; ++tm)
#pragma unroll
        for (int tn = 0; tn < 4; ++tn)
          acc[tm][tn] = __builtin_amdgcn_wmma_f32_16x16x32_f16(
              false, af[tm], false, bf[tn], (short)0, acc[tm][tn], false, false);
    }

    __syncthreads();                       // all waves done reading buf (s&1)
    if (s + 2 < NSTAGE) issue_stage(s + 2, s & 1);
  }

  // Epilogue: C/D layout -> VGPR r holds M = r + 8*hi, N = lo.
  // Only j > i consumed; every such element feeds row i and mirrored col j.
  // Fixed-shift logsumexp: sim in [-1/T, 1/T] so exp(sim - 1/T) <= 1 always.
#pragma unroll
  for (int tm = 0; tm < 4; ++tm) {
#pragma unroll
    for (int tn = 0; tn < 4; ++tn) {
      const int nloc = waveN * 64 + tn * 16 + lo;
      const int j    = colBase + nloc;
      const int lj   = lCol[nloc];
#pragma unroll
      for (int r = 0; r < 8; ++r) {
        const int m = waveM * 64 + tm * 16 + r + 8 * hi;
        const int i = rowBase + m;
        if (j <= i) continue;              // below/on diagonal: not consumed
        const float sv = acc[tm][tn][r] * INV_T;
        const float e  = __expf(sv - INV_T);
        atomicAdd(&sSum[m], e);
        atomicAdd(&cSum[nloc], e);
        if (lRow[m] == lj) {
          atomicAdd(&sPos[m], sv); atomicAdd(&sCnt[m], 1);
          atomicAdd(&cPos[nloc], sv); atomicAdd(&cCnt[nloc], 1);
        }
      }
    }
  }
  __syncthreads();

  atomicAdd(&gSum[rowBase + tid], sSum[tid]);
  atomicAdd(&gPos[rowBase + tid], sPos[tid]);
  atomicAdd(&gCnt[rowBase + tid], sCnt[tid]);
  if (tid < BN) {
    atomicAdd(&gSum[colBase + tid], cSum[tid]);
    atomicAdd(&gPos[colBase + tid], cPos[tid]);
    atomicAdd(&gCnt[colBase + tid], cCnt[tid]);
  }
}

// ---------------------------------------------------------------------------
// Kernel 4: final scalar reduction
// per_row = log(sumexp) + 1/T - pos_sum/cnt  (rows with cnt==0 contribute 0)
// ---------------------------------------------------------------------------
__global__ __launch_bounds__(256)
void scl_finish(const float* __restrict__ gSum, const float* __restrict__ gPos,
                const int* __restrict__ gCnt, float* __restrict__ out) {
  __shared__ float red[256];
  const int tid = threadIdx.x;
  float acc = 0.f;
  for (int i = tid; i < NROWS; i += 256) {
    const int c = gCnt[i];
    if (c > 0) acc += logf(gSum[i]) + INV_T - gPos[i] / (float)c;
  }
  red[tid] = acc;
  __syncthreads();
  for (int s = 128; s > 0; s >>= 1) {
    if (tid < s) red[tid] += red[tid + s];
    __syncthreads();
  }
  if (tid == 0) out[0] = red[0];
}

// ---------------------------------------------------------------------------
extern "C" void kernel_launch(void* const* d_in, const int* in_sizes, int n_in,
                              void* d_out, int out_size, void* d_ws,
                              size_t ws_size, hipStream_t stream) {
  const float* vecs   = (const float*)d_in[0];
  const int*   labels = (const int*)d_in[1];
  float*       out    = (float*)d_out;

  // Workspace layout: [f16 Anorm 16MB][gSum 32KB][gPos 32KB][gCnt 32KB]
  _Float16* anorm = (_Float16*)d_ws;
  float* gSum = (float*)((char*)d_ws + (size_t)NROWS * DIMS * sizeof(_Float16));
  float* gPos = gSum + NROWS;
  int*   gCnt = (int*)(gPos + NROWS);

  scl_normalize<<<NROWS, 256, 0, stream>>>(vecs, anorm);
  scl_zero<<<NROWS / 256, 256, 0, stream>>>(gSum, gPos, gCnt);
  dim3 grid(NROWS / BN, NROWS / BM);   // 64 x 32
  scl_tiles<<<grid, 256, 0, stream>>>(anorm, labels, gSum, gPos, gCnt);
  scl_finish<<<1, 256, 0, stream>>>(gSum, gPos, gCnt, out);
}
